// MovePredictor_22514218566318
// MI455X (gfx1250) — matrix operand order repaired and checked by reading, not compiled, not run on hardware
//
#include <hip/hip_runtime.h>
#include <hip/hip_bf16.h>

typedef _Float16 f16;
typedef __attribute__((ext_vector_type(16))) _Float16 v16h;
typedef __attribute__((ext_vector_type(8)))  _Float16 v8h;
typedef __attribute__((ext_vector_type(4)))  _Float16 v4h;
typedef __attribute__((ext_vector_type(8)))  float    v8f;
typedef __attribute__((ext_vector_type(4)))  float    v4f;

#define N_NODES  100000
#define N_EDGES  1600000
#define N_MOVES  200000
#define DIM      256          // in_channels
#define IN1      512          // 2*DIM
#define HID1     512
#define HID2     256

#define TILE_M   64           // moves per block; 200000/64 = 3125 exactly
#define RT       (TILE_M/16)  // 4 row-tiles
#define XS_STRIDE 520         // halves; 1040B pitch: 16B aligned, bank-skewed
#define H1_STRIDE 520
#define H2_STRIDE 264

// ---------------- weight f32 -> f16 conversion (one-time per launch) -------
__global__ void cvt_weights_kernel(const float* __restrict__ W1,
                                   const float* __restrict__ W2,
                                   f16* __restrict__ W1h, f16* __restrict__ W2h) {
  int i = blockIdx.x * blockDim.x + threadIdx.x;
  if (i < HID1 * IN1)  W1h[i] = (f16)W1[i];
  if (i < HID2 * HID1) W2h[i] = (f16)W2[i];
}

// ------------- node table f32 -> f16 (vectorized, once per launch) ---------
__global__ void cvt_emb_kernel(const float* __restrict__ emb,
                               f16* __restrict__ embh) {
  long i = ((long)blockIdx.x * blockDim.x + threadIdx.x) * 4;
  if (i < (long)N_NODES * DIM) {
    v4f v = *(const v4f*)(emb + i);
    v4h h; h.x = (f16)v.x; h.y = (f16)v.y; h.z = (f16)v.z; h.w = (f16)v.w;
    *(v4h*)(embh + i) = h;
  }
}

// ---- A fragment (16x32 f16) from LDS, CDNA5 WMMA layout -------------------
// lanes 0-15: row M=lane, K in [kb .. kb+7] and [kb+16 .. kb+23]
// lanes16-31: row M=lane-16, K in [kb+8 .. kb+15] and [kb+24 .. kb+31]
__device__ __forceinline__ v16h load_a_lds(const f16* base, int stride,
                                           int row0, int kbase, int lane) {
  int row = row0 + (lane & 15);
  int hi  = (lane >> 4) * 8;
  const f16* p = base + row * stride + kbase + hi;
  v8h lo = *(const v8h*)p;           // K: kbase+hi    .. +7
  v8h hh = *(const v8h*)(p + 16);    // K: kbase+16+hi .. +7
  return __builtin_shufflevector(lo, hh, 0,1,2,3,4,5,6,7,8,9,10,11,12,13,14,15);
}

// ---- B fragment (32x16 f16) from global row-major W[out][in] --------------
// lanes 0-15: col N=lane, K in [kb .. kb+7] and [kb+16 .. kb+23]; lanes16-31: K in [kb+8 .. kb+15] and [kb+24 .. kb+31]
__device__ __forceinline__ v16h load_b_glb(const f16* __restrict__ W, int ldw,
                                           int ntile, int kbase, int lane) {
  int n  = ntile * 16 + (lane & 15);
  int hi = (lane >> 4) * 8;
  const f16* p = W + (long)n * ldw + kbase + hi;
  v8h lo = *(const v8h*)p;
  v8h hh = *(const v8h*)(p + 16);
  return __builtin_shufflevector(lo, hh, 0,1,2,3,4,5,6,7,8,9,10,11,12,13,14,15);
}

// ---------------------------- fused MLP ------------------------------------
__launch_bounds__(256)
__global__ void mlp_fused_kernel(const float* __restrict__ emb,
                                 const f16*   __restrict__ embh, // null -> f32 path
                                 const int*   __restrict__ edge_index,
                                 const int*   __restrict__ moves,
                                 const f16*   __restrict__ W1h,
                                 const f16*   __restrict__ W2h,
                                 const float* __restrict__ b1,
                                 const float* __restrict__ b2,
                                 const float* __restrict__ W3,
                                 const float* __restrict__ b3,
                                 float* __restrict__ out) {
  // ~131 KB static LDS (CDNA5: up to 320 KB per workgroup)
  __shared__ __align__(16) f16 xs [TILE_M * XS_STRIDE]; // X tile, reused as h2
  __shared__ __align__(16) f16 h1s[TILE_M * H1_STRIDE]; // hidden1 tile
  __shared__ float red[256];

  const int t    = threadIdx.x;
  const int lane = t & 31;
  const int wave = t >> 5;
  const int m0   = blockIdx.x * TILE_M;

  // warm L2 for this block's weight stripes (global_prefetch_b8)
  __builtin_prefetch(W1h + (size_t)(wave * 4 * 16 + (lane & 15)) * IN1, 0, 1);
  __builtin_prefetch(W2h + (size_t)(wave * 2 * 16 + (lane & 15)) * HID1, 0, 1);

  // ---------- stage 0: gather embeddings into LDS ----------
  {
    const int r = t >> 2;           // 4 threads per row, 64 rows
    const int q = t & 3;
    int m = m0 + r; if (m >= N_MOVES) m = N_MOVES - 1;
    const int e = moves[m];
    const int s = edge_index[e];
    const int g = edge_index[N_EDGES + e];
    if (embh) {
      // fast path: f16 table, async DMA straight into LDS (ASYNCcnt)
      const f16* srow = embh + (long)s * DIM;
      const f16* trow = embh + (long)g * DIM;
#pragma unroll
      for (int it = 0; it < 16; ++it) {
        int c = q + it * 4;         // 16B chunk id, 0..63 (1024B per row)
        const f16* gp = (c < 32) ? (srow + c * 8) : (trow + (c - 32) * 8);
        unsigned lds = (unsigned)(uintptr_t)(&xs[r * XS_STRIDE + c * 8]);
        asm volatile("global_load_async_to_lds_b128 %0, %1, off"
                     :: "v"(lds), "v"((unsigned long long)(uintptr_t)gp)
                     : "memory");
      }
      asm volatile("s_wait_asynccnt 0" ::: "memory");
    } else {
      // fallback: gather f32 rows, convert in VALU
      const float* srow = emb + (long)s * DIM;
      const float* trow = emb + (long)g * DIM;
#pragma unroll
      for (int it = 0; it < 32; ++it) {
        int c = q + it * 4;         // float4 chunk id, 0..127
        const float* p = (c < 64) ? (srow + c * 4) : (trow + (c - 64) * 4);
        v4f v = *(const v4f*)p;
        v4h h; h.x = (f16)v.x; h.y = (f16)v.y; h.z = (f16)v.z; h.w = (f16)v.w;
        *(v4h*)&xs[r * XS_STRIDE + c * 4] = h;
      }
    }
  }
  __syncthreads();

  // ---------- layer 1: h1 = relu(X @ W1^T + b1), [64x512] ----------
  // wave owns 4 n-tiles x 4 r-tiles: each B fragment feeds 4 WMMAs.
  {
    const int nbase = wave * 4;     // 32 n-tiles / 8 waves
    v8f acc[4][4];                  // [n][r]
#pragma unroll
    for (int i = 0; i < 4; ++i) {
      float bv = b1[(nbase + i) * 16 + (lane & 15)];
#pragma unroll
      for (int r = 0; r < RT; ++r)
#pragma unroll
        for (int j = 0; j < 8; ++j) acc[i][r][j] = bv;
    }
#pragma unroll 2
    for (int ks = 0; ks < 16; ++ks) {
      v16h a[RT];
#pragma unroll
      for (int r = 0; r < RT; ++r)
        a[r] = load_a_lds(xs, XS_STRIDE, r * 16, ks * 32, lane);
#pragma unroll
      for (int i = 0; i < 4; ++i) {
        v16h b = load_b_glb(W1h, IN1, nbase + i, ks * 32, lane);
#pragma unroll
        for (int r = 0; r < RT; ++r)
          acc[i][r] = __builtin_amdgcn_wmma_f32_16x16x32_f16(
              false, a[r], false, b, (short)0, acc[i][r], false, false);
      }
    }
    const int rb = (lane >> 4) * 8;  // C/D layout: lanes>=16 hold M=8..15
#pragma unroll
    for (int i = 0; i < 4; ++i) {
      int col = (nbase + i) * 16 + (lane & 15);
#pragma unroll
      for (int r = 0; r < RT; ++r)
#pragma unroll
        for (int j = 0; j < 8; ++j) {
          float v = acc[i][r][j]; v = v > 0.f ? v : 0.f;
          h1s[(r * 16 + rb + j) * H1_STRIDE + col] = (f16)v;
        }
    }
  }
  __syncthreads();

  // ---------- layer 2: h2 = relu(h1 @ W2^T + b2), [64x256] ----------
  {
    const int nbase = wave * 2;     // 16 n-tiles / 8 waves
    v8f acc[2][4];
#pragma unroll
    for (int i = 0; i < 2; ++i) {
      float bv = b2[(nbase + i) * 16 + (lane & 15)];
#pragma unroll
      for (int r = 0; r < RT; ++r)
#pragma unroll
        for (int j = 0; j < 8; ++j) acc[i][r][j] = bv;
    }
#pragma unroll 2
    for (int ks = 0; ks < 16; ++ks) {
      v16h a[RT];
#pragma unroll
      for (int r = 0; r < RT; ++r)
        a[r] = load_a_lds(h1s, H1_STRIDE, r * 16, ks * 32, lane);
#pragma unroll
      for (int i = 0; i < 2; ++i) {
        v16h b = load_b_glb(W2h, HID1, nbase + i, ks * 32, lane);
#pragma unroll
        for (int r = 0; r < RT; ++r)
          acc[i][r] = __builtin_amdgcn_wmma_f32_16x16x32_f16(
              false, a[r], false, b, (short)0, acc[i][r], false, false);
      }
    }
    f16* h2s = xs;                  // xs is dead after layer 1 (synced)
    const int rb = (lane >> 4) * 8;
#pragma unroll
    for (int i = 0; i < 2; ++i) {
      int col = (nbase + i) * 16 + (lane & 15);
#pragma unroll
      for (int r = 0; r < RT; ++r)
#pragma unroll
        for (int j = 0; j < 8; ++j) {
          float v = acc[i][r][j]; v = v > 0.f ? v : 0.f;
          h2s[(r * 16 + rb + j) * H2_STRIDE + col] = (f16)v;
        }
    }
  }
  __syncthreads();

  // ---------- layer 3: logits = h2 @ W3^T + b3  (tiny, VALU) ----------
  {
    const f16* h2s = xs;
    const int row = t >> 2, part = t & 3;   // 4 partials per row, 64 rows
    const f16* hp = h2s + row * H2_STRIDE + part * 64;
    float s = 0.f;
#pragma unroll
    for (int k = 0; k < 64; ++k) s += (float)hp[k] * W3[part * 64 + k];
    red[t] = s;
    __syncthreads();
    if (t < TILE_M) {
      float tot = b3[0];
#pragma unroll
      for (int p = 0; p < 4; ++p) tot += red[t * 4 + p];
      int m = m0 + t;
      if (m < N_MOVES) out[m] = tot;
    }
  }
}

extern "C" void kernel_launch(void* const* d_in, const int* in_sizes, int n_in,
                              void* d_out, int out_size, void* d_ws, size_t ws_size,
                              hipStream_t stream) {
  const float* emb = (const float*)d_in[0];
  const int*   ei  = (const int*)  d_in[1];
  const int*   mv  = (const int*)  d_in[2];
  const float* W1  = (const float*)d_in[3];
  const float* b1  = (const float*)d_in[4];
  const float* W2  = (const float*)d_in[5];
  const float* b2  = (const float*)d_in[6];
  const float* W3  = (const float*)d_in[7];
  const float* b3  = (const float*)d_in[8];
  float* out = (float*)d_out;

  f16* W1h = (f16*)d_ws;                              // 512*512 halves
  f16* W2h = W1h + HID1 * IN1;                        // 256*512 halves
  f16* embh = W2h + HID2 * HID1;                      // 100000*256 halves
  const size_t need = (size_t)(HID1 * IN1 + HID2 * HID1) * 2 +
                      (size_t)N_NODES * DIM * 2;
  const bool have_tbl = ws_size >= need;

  cvt_weights_kernel<<<(HID1 * IN1 + 255) / 256, 256, 0, stream>>>(W1, W2, W1h, W2h);
  if (have_tbl) {
    long n4 = (long)N_NODES * DIM / 4;
    cvt_emb_kernel<<<(unsigned)((n4 + 255) / 256), 256, 0, stream>>>(emb, embh);
  }

  mlp_fused_kernel<<<(N_MOVES + TILE_M - 1) / TILE_M, 256, 0, stream>>>(
      emb, have_tbl ? embh : (const f16*)nullptr, ei, mv,
      W1h, W2h, b1, b2, W3, b3, out);
}
